// DiscreteOptionActor_3582002725160
// MI455X (gfx1250) — matrix operand order repaired and checked by reading, not compiled, not run on hardware
//
#include <hip/hip_runtime.h>
#include <hip/hip_bf16.h>

// ---------------------------------------------------------------------------
// DiscreteOptionActor on MI455X (gfx1250): per-option 3-layer MLP, computing
// ONLY the selected option trunk per sample via option-bucketing + f16 WMMA.
// ---------------------------------------------------------------------------

typedef _Float16 v16h __attribute__((ext_vector_type(16)));
typedef _Float16 v8h  __attribute__((ext_vector_type(8)));
typedef float    v8f  __attribute__((ext_vector_type(8)));

#define OBS_DIM 128
#define ACT_DIM 18
#define OPT_DIM 16
#define H1 256
#define H2 256
#define BATCH 65536
#define TM 64            // rows (samples) per workgroup tile

// packed-weight element counts (f16)
#define W1P_ELEMS (OPT_DIM * 4  * 16 * 512)   // K=128 -> 4 kt, N=256 -> 16 nt
#define W2P_ELEMS (OPT_DIM * 8  * 16 * 512)   // K=256 -> 8 kt, N=256 -> 16 nt
#define W3P_ELEMS (OPT_DIM * 8  * 2  * 512)   // K=256 -> 8 kt, N=32(pad) -> 2 nt

// ---------------------------------------------------------------------------
// Prep kernels
// ---------------------------------------------------------------------------
__global__ void zero_ints(int* p) {
  if (threadIdx.x < 48) p[threadIdx.x] = 0;
}

// Repack fp32 weight [O][K][N] into f16 WMMA-B lane-major blocks.
// dst index = blk*512 + lane*16 + e, blk = (o*nnt + nt)*nkt + kt.
// B-operand (32x16, KxN) 16-bit layout: lane = N column (0..15 twice),
// lanes 0-15: e0..7 -> K 0..7, e8..15 -> K 16..23; lanes 16-31: e0..7 -> K 8..15, e8..15 -> K 24..31.
__global__ void pack_weights(const float* __restrict__ src, _Float16* __restrict__ dst,
                             int K, int N, int nkt, int nnt, int total) {
  int tid = blockIdx.x * blockDim.x + threadIdx.x;
  if (tid >= total) return;
  int e   = tid & 15;
  int l   = (tid >> 4) & 31;
  int blk = tid >> 9;
  int kt  = blk % nkt;
  int tmp = blk / nkt;
  int nt  = tmp % nnt;
  int o   = tmp / nnt;
  int n = (nt << 4) + (l & 15);
  int k = (kt << 5) + ((l < 16) ? 0 : 8) + ((e < 8) ? e : e + 8);
  float v = 0.0f;
  if (n < N) v = src[(o * K + k) * N + n];
  dst[tid] = (_Float16)v;
}

__global__ void count_opts(const int* __restrict__ opt, int* __restrict__ counts, int n) {
  int i = blockIdx.x * blockDim.x + threadIdx.x;
  if (i < n) atomicAdd(&counts[opt[i] & 15], 1);
}

__global__ void scan16(const int* __restrict__ counts, int* __restrict__ offsets,
                       int* __restrict__ cursor) {
  if (threadIdx.x == 0) {
    int s = 0;
    for (int i = 0; i < 16; ++i) { offsets[i] = s; cursor[i] = s; s += counts[i]; }
  }
}

__global__ void scatter_perm(const int* __restrict__ opt, int* __restrict__ cursor,
                             int* __restrict__ perm, int n) {
  int i = blockIdx.x * blockDim.x + threadIdx.x;
  if (i < n) {
    int p = atomicAdd(&cursor[opt[i] & 15], 1);
    perm[p] = i;
  }
}

// ---------------------------------------------------------------------------
// A-operand assembly from row-major f16 LDS buffer.
// 16-bit A (16x32) layout: lane = M row (0..15 twice);
// lanes 0-15:  e0..7 -> K 0..7,  e8..15 -> K 16..23
// lanes 16-31: e0..7 -> K 8..15, e8..15 -> K 24..31
// ---------------------------------------------------------------------------
__device__ __forceinline__ v16h load_A(const _Float16* buf, int width, int mt, int kt, int lane) {
  int row = mt * 16 + (lane & 15);
  int ks  = (kt << 5) + ((lane < 16) ? 0 : 8);
  union { v16h v; v8h h[2]; } u;
  u.h[0] = *(const v8h*)(buf + row * width + ks);
  u.h[1] = *(const v8h*)(buf + row * width + ks + 16);
  return u.v;
}

// ---------------------------------------------------------------------------
// Main fused MLP kernel: one workgroup = (option o, 64-sample tile).
// 8 waves: wave w -> M-tile (w>>1), N-tile half (w&1) for layers 1/2;
// layer 3: M-tile (w>>1), N-tile (w&1).
// ---------------------------------------------------------------------------
__global__ __launch_bounds__(256) void actor_mlp_wmma(
    const float* __restrict__ obs,
    const float* __restrict__ b1, const float* __restrict__ b2, const float* __restrict__ b3,
    const _Float16* __restrict__ W1p, const _Float16* __restrict__ W2p,
    const _Float16* __restrict__ W3p,
    const int* __restrict__ counts, const int* __restrict__ offsets,
    const int* __restrict__ perm,
    float* __restrict__ out)
{
  int o     = blockIdx.x;
  int n_o   = counts[o];
  int start = blockIdx.y * TM;
  if (start >= n_o) return;                 // uniform across the block
  int valid = n_o - start; if (valid > TM) valid = TM;
  int base  = offsets[o] + start;

  // sX: first TM*128 holds obs(f16), later reused as h2 (TM*256). 32KB.
  __shared__ _Float16 sX[TM * 256];
  __shared__ _Float16 sH1[TM * 256];        // 32KB  -> total exactly 64KB

  int t = threadIdx.x;

  // gather obs rows via perm, convert to f16 (zero-pad invalid rows)
  for (int idx = t; idx < TM * OBS_DIM; idx += 256) {
    int r = idx >> 7, c = idx & 127;
    float v = 0.0f;
    if (r < valid) v = obs[perm[base + r] * OBS_DIM + c];
    sX[idx] = (_Float16)v;
  }
  __syncthreads();

  int wave  = t >> 5, lane = t & 31;
  int mt    = wave >> 1;                    // 0..3
  int rbase = (lane < 16) ? 0 : 8;          // C/D: lanes 0-15 rows 0-7, 16-31 rows 8-15
  int cit   = lane & 15;                    // column within 16-wide N tile

  // ------------------- layer 1: X[64x128] * W1[128x256] -> H1 ---------------
  {
    v16h a[4];
#pragma unroll
    for (int kt = 0; kt < 4; ++kt) a[kt] = load_A(sX, OBS_DIM, mt, kt, lane);
    int ntBase = (wave & 1) * 8;
#pragma unroll
    for (int j = 0; j < 8; ++j) {
      int nt = ntBase + j;
      const _Float16* wb = W1p + (((o * 16 + nt) * 4) << 9) + (lane << 4);
      v8f acc = {};
#pragma unroll
      for (int kt = 0; kt < 4; ++kt) {
        v16h b = *(const v16h*)(wb + (kt << 9));
        acc = __builtin_amdgcn_wmma_f32_16x16x32_f16(false, a[kt], false, b,
                                                     (short)0, acc, false, false);
      }
      float bias = b1[o * H1 + nt * 16 + cit];
      int col = nt * 16 + cit;
#pragma unroll
      for (int r = 0; r < 8; ++r) {
        float h = acc[r] + bias;
        h = h > 0.0f ? h : 0.0f;
        sH1[(mt * 16 + rbase + r) * H1 + col] = (_Float16)h;
      }
    }
  }
  __syncthreads();   // H1 write->read; also X read -> X(=h2) write

  // ------------------- layer 2: H1[64x256] * W2[256x256] -> X(h2) -----------
  {
    v16h a[8];
#pragma unroll
    for (int kt = 0; kt < 8; ++kt) a[kt] = load_A(sH1, H1, mt, kt, lane);
    int ntBase = (wave & 1) * 8;
#pragma unroll
    for (int j = 0; j < 8; ++j) {
      int nt = ntBase + j;
      const _Float16* wb = W2p + (((o * 16 + nt) * 8) << 9) + (lane << 4);
      v8f acc = {};
#pragma unroll
      for (int kt = 0; kt < 8; ++kt) {
        v16h b = *(const v16h*)(wb + (kt << 9));
        acc = __builtin_amdgcn_wmma_f32_16x16x32_f16(false, a[kt], false, b,
                                                     (short)0, acc, false, false);
      }
      float bias = b2[o * H2 + nt * 16 + cit];
      int col = nt * 16 + cit;
#pragma unroll
      for (int r = 0; r < 8; ++r) {
        float h = acc[r] + bias;
        h = h > 0.0f ? h : 0.0f;
        sX[(mt * 16 + rbase + r) * H2 + col] = (_Float16)h;
      }
    }
  }
  __syncthreads();

  // ------------------- layer 3: H2[64x256] * W3[256x18pad32] -> out ---------
  {
    v16h a[8];
#pragma unroll
    for (int kt = 0; kt < 8; ++kt) a[kt] = load_A(sX, H2, mt, kt, lane);
    int nt = wave & 1;                       // 0..1 (N padded to 32)
    const _Float16* wb = W3p + (((o * 2 + nt) * 8) << 9) + (lane << 4);
    v8f acc = {};
#pragma unroll
    for (int kt = 0; kt < 8; ++kt) {
      v16h b = *(const v16h*)(wb + (kt << 9));
      acc = __builtin_amdgcn_wmma_f32_16x16x32_f16(false, a[kt], false, b,
                                                   (short)0, acc, false, false);
    }
    int col = nt * 16 + cit;
    if (col < ACT_DIM) {
      float bias = b3[o * ACT_DIM + col];
#pragma unroll
      for (int r = 0; r < 8; ++r) {
        int row = mt * 16 + rbase + r;
        if (row < valid) out[perm[base + row] * ACT_DIM + col] = acc[r] + bias;
      }
    }
  }
}

// ---------------------------------------------------------------------------
// Host launcher (graph-capture safe: all work on `stream`, scratch in d_ws)
// ---------------------------------------------------------------------------
extern "C" void kernel_launch(void* const* d_in, const int* in_sizes, int n_in,
                              void* d_out, int out_size, void* d_ws, size_t ws_size,
                              hipStream_t stream) {
  const float* obs    = (const float*)d_in[0];
  const int*   option = (const int*)d_in[1];
  const float* W1     = (const float*)d_in[2];
  const float* b1     = (const float*)d_in[3];
  const float* W2     = (const float*)d_in[4];
  const float* b2     = (const float*)d_in[5];
  const float* W3     = (const float*)d_in[6];
  const float* b3     = (const float*)d_in[7];
  float*       out    = (float*)d_out;

  // workspace layout
  _Float16* wsh = (_Float16*)d_ws;
  _Float16* W1p = wsh;
  _Float16* W2p = wsh + W1P_ELEMS;
  _Float16* W3p = wsh + W1P_ELEMS + W2P_ELEMS;
  int* wsi     = (int*)(wsh + (W1P_ELEMS + W2P_ELEMS + W3P_ELEMS));
  int* counts  = wsi;
  int* offsets = wsi + 16;
  int* cursor  = wsi + 32;
  int* perm    = wsi + 48;

  zero_ints<<<1, 64, 0, stream>>>(wsi);

  pack_weights<<<(W1P_ELEMS + 255) / 256, 256, 0, stream>>>(W1, W1p, OBS_DIM, H1, 4, 16, W1P_ELEMS);
  pack_weights<<<(W2P_ELEMS + 255) / 256, 256, 0, stream>>>(W2, W2p, H1, H2, 8, 16, W2P_ELEMS);
  pack_weights<<<(W3P_ELEMS + 255) / 256, 256, 0, stream>>>(W3, W3p, H2, ACT_DIM, 8, 2, W3P_ELEMS);

  count_opts<<<BATCH / 256, 256, 0, stream>>>(option, counts, BATCH);
  scan16<<<1, 32, 0, stream>>>(counts, offsets, cursor);
  scatter_perm<<<BATCH / 256, 256, 0, stream>>>(option, cursor, perm, BATCH);

  dim3 grid(OPT_DIM, BATCH / TM);   // y tiles generous; empty tiles exit early
  actor_mlp_wmma<<<grid, 256, 0, stream>>>(obs, b1, b2, b3, W1p, W2p, W3p,
                                           counts, offsets, perm, out);
}
